// LSTM_Stack_50208167690396
// MI455X (gfx1250) — compile-verified, matrix operand order checked by reading, not run on hardware
//
#include <hip/hip_runtime.h>

typedef __attribute__((ext_vector_type(8)))  float   v8f;
typedef __attribute__((ext_vector_type(16))) __bf16  v16bf;
typedef __attribute__((ext_vector_type(8)))  __bf16  v8bf;

#define MAXT      129
#define BATCH     256
#define DIN       128
#define DS        64
#define HID       1024
#define ZDIM      (DIN + DS)          // 192
#define KCAT      (ZDIM + HID)        // 1216
#define G4H       (4 * HID)           // 4096
#define NOUT      194                 // Din + Ds + 2
#define NOUTP     208                 // padded to 13*16
#define ENC_STEPS 65                  // k + 2, k = 63

// ---- gate-GEMM tiling (LDS double-buffered, async loads) ----
#define KC      64                    // K chunk
#define NCH     (KCAT / KC)           // 19
#define BM      128                   // block M tile (8 waves x 16)
#define BN      64                    // block N tile (4 x 16)
#define LPAD    8                     // LDS row pad (elements) -> 144B stride
#define LROW    (KC + LPAD)           // 72
#define A_ELE   (BM * LROW)           // 9216
#define W_ELE   (BN * LROW)           // 4608
#define BUF_ELE (A_ELE + W_ELE)       // 13824 bf16 per buffer (27.6 KB)

static __device__ __forceinline__ float sigm(float x) { return 1.0f / (1.0f + __expf(-x)); }

static __device__ __forceinline__ __bf16 f2bf(float f) {
    unsigned u = __builtin_bit_cast(unsigned, f);
    unsigned r = u + 0x7FFFu + ((u >> 16) & 1u);      // round-to-nearest-even
    unsigned short h = (unsigned short)(r >> 16);
    return __builtin_bit_cast(__bf16, h);
}

union FragBF { v16bf v16; v8bf v8[2]; };

static __device__ __forceinline__ void async_b128(unsigned ldsOff, unsigned long long gaddr) {
    // GLOBAL_LOAD_ASYNC_TO_LDS_B128 (GV form): VDST = LDS byte offset, VADDR = 64-bit global.
    asm volatile("global_load_async_to_lds_b128 %0, %1, off" :: "v"(ldsOff), "v"(gaddr) : "memory");
}
static __device__ __forceinline__ void wait_async0() {
    asm volatile("s_wait_asynccnt 0x0" ::: "memory");
}

// ---------------------------------------------------------------- prep kernels
__global__ void k_zero_f32(float* __restrict__ p, int n) {
    int i = blockIdx.x * blockDim.x + threadIdx.x;
    if (i < n) p[i] = 0.0f;
}

// Wcat[n][0:192] = w_ih[n], Wcat[n][192:1216] = w_hh[n]   (bf16)
__global__ void k_build_wcat(__bf16* __restrict__ dst, const float* __restrict__ wih,
                             const float* __restrict__ whh) {
    int i = blockIdx.x * blockDim.x + threadIdx.x;
    if (i >= G4H * KCAT) return;
    int n = i / KCAT, kc = i % KCAT;
    float v = (kc < ZDIM) ? wih[n * ZDIM + kc] : whh[n * HID + (kc - ZDIM)];
    dst[i] = f2bf(v);
}

// padded (208 x 1024) bf16 w_ro + padded f32 bias
__global__ void k_build_wro(__bf16* __restrict__ dst, float* __restrict__ bdst,
                            const float* __restrict__ wro, const float* __restrict__ bro) {
    int i = blockIdx.x * blockDim.x + threadIdx.x;
    if (i >= NOUTP * HID) return;
    int n = i / HID, kk = i % HID;
    dst[i] = f2bf((n < NOUT) ? wro[n * HID + kk] : 0.0f);
    if (i < NOUTP) bdst[i] = (i < NOUT) ? bro[i] : 0.0f;
}

__global__ void k_build_bias(float* __restrict__ dst, const float* __restrict__ bih,
                             const float* __restrict__ bhh) {
    int i = blockIdx.x * blockDim.x + threadIdx.x;
    if (i < G4H) dst[i] = bih[i] + bhh[i];
}

// A-row layout per batch b: [ x (128) | rt (64) | h (1024) ] in bf16
__global__ void k_init_abuf(__bf16* __restrict__ A, const float* __restrict__ x0) {
    int i = blockIdx.x * blockDim.x + threadIdx.x;
    if (i >= BATCH * KCAT) return;
    int b = i / KCAT, d = i % KCAT;
    A[i] = f2bf((d < DIN) ? x0[b * DIN + d] : 0.0f);
}

// ---------------------------------------------------------------- gate GEMM
// gates(256 x 4096) = A(256 x 1216) * Wcat^T, +bias.
// Block = 8 waves, tile BM=128 x BN=64; K chunks of 64 staged into LDS with
// GLOBAL_LOAD_ASYNC_TO_LDS_B128 (ASYNCcnt), double buffered.
__global__ __launch_bounds__(256) void k_gemm_gates(const __bf16* __restrict__ A,
                                                    const __bf16* __restrict__ W,
                                                    const float* __restrict__ bias,
                                                    float* __restrict__ gates) {
    __shared__ __align__(16) __bf16 smem[2 * BUF_ELE];   // 55 KB of 320 KB/WGP

    const int tid  = threadIdx.x;
    const int lane = tid & 31;
    const int wid  = tid >> 5;                 // 0..7 -> wave's 16-row M sub-tile
    const int bm0  = (blockIdx.x & 1) * BM;    // 2 M-blocks
    const int bn0  = (blockIdx.x >> 1) * BN;   // 64 N-blocks

    const unsigned ldsBase = (unsigned)(size_t)(void*)smem;   // LDS byte offset
    const unsigned long long aG = (unsigned long long)(size_t)A + (size_t)bm0 * (KCAT * 2);
    const unsigned long long wG = (unsigned long long)(size_t)W + (size_t)bn0 * (KCAT * 2);

    // stage K-chunk c into buffer `buf`: A 128x64 (1024 segs) + W 64x64 (512 segs), 6/thread
    auto stage = [&](int c, int buf) {
        const unsigned bufA = ldsBase + (unsigned)buf * (BUF_ELE * 2);
        const unsigned bufW = bufA + A_ELE * 2;
        const unsigned long long kByte = (unsigned long long)c * (KC * 2);
#pragma unroll
        for (int i = 0; i < 4; ++i) {              // A segments 0..1023
            int s = tid + i * 256;
            int r = s >> 3, q = s & 7;
            async_b128(bufA + (unsigned)(r * (LROW * 2) + q * 16),
                       aG + (unsigned long long)r * (KCAT * 2) + kByte + (unsigned)(q * 16));
        }
#pragma unroll
        for (int i = 0; i < 2; ++i) {              // W segments 0..511
            int s = tid + i * 256;
            int r = s >> 3, q = s & 7;
            async_b128(bufW + (unsigned)(r * (LROW * 2) + q * 16),
                       wG + (unsigned long long)r * (KCAT * 2) + kByte + (unsigned)(q * 16));
        }
    };

    const int row  = lane & 15;
    const int hi16 = lane >> 4;

    v8f acc[4] = {v8f{}, v8f{}, v8f{}, v8f{}};

    stage(0, 0);
    wait_async0();
    __syncthreads();

    for (int c = 0; c < NCH; ++c) {
        if (c + 1 < NCH) stage(c + 1, (c + 1) & 1);      // prefetch next chunk

        const __bf16* aT = smem + (c & 1) * BUF_ELE;
        const __bf16* wT = aT + A_ELE;
#pragma unroll
        for (int ks = 0; ks < 2; ++ks) {                 // two K=32 steps per chunk
            FragBF af;
            const __bf16* ap = aT + (wid * 16 + row) * LROW + ks * 32 + hi16 * 8;
            af.v8[0] = *(const v8bf*)(ap);
            af.v8[1] = *(const v8bf*)(ap + 16);
#pragma unroll
            for (int j = 0; j < 4; ++j) {
                const __bf16* wp = wT + (j * 16 + row) * LROW + ks * 32 + hi16 * 8;
                FragBF bf;
                bf.v8[0] = *(const v8bf*)(wp);
                bf.v8[1] = *(const v8bf*)(wp + 16);
                acc[j] = __builtin_amdgcn_wmma_f32_16x16x32_bf16(
                    false, af.v16, false, bf.v16, (short)0, acc[j], false, false);
            }
        }
        wait_async0();          // latency of chunk c+1 hidden behind the 8 WMMAs above
        __syncthreads();
    }

    int m0 = bm0 + wid * 16 + hi16 * 8;
    int nc = lane & 15;
#pragma unroll
    for (int j = 0; j < 4; ++j) {
        int n = bn0 + j * 16 + nc;
        float bv = bias[n];
#pragma unroll
        for (int r = 0; r < 8; ++r)
            gates[(size_t)(m0 + r) * G4H + n] = acc[j][r] + bv;
    }
}

// out(256 x 208) = h(256 x 1024) * w_ro^T, +bias.  One wave -> 16x16 tile. 208 waves.
__global__ __launch_bounds__(128) void k_gemm_out(const __bf16* __restrict__ A,
                                                  const __bf16* __restrict__ W,
                                                  const float* __restrict__ bias,
                                                  float* __restrict__ outB) {
    int wave = blockIdx.x * 4 + (threadIdx.x >> 5);
    int lane = threadIdx.x & 31;
    int mt = wave & 15;        // 16 M-tiles
    int nt = wave >> 4;        // 13 N-tiles
    int row  = lane & 15;
    int koff = (lane >> 4) * 8;

    const __bf16* aRow = A + (size_t)(mt * 16 + row) * KCAT + ZDIM;   // h region
    const __bf16* wRow = W + (size_t)(nt * 16 + row) * HID;

    v8f acc = {};
    for (int kk = 0; kk < HID; kk += 32) {
        FragBF af, bf;
        af.v8[0] = *(const v8bf*)(aRow + kk + koff);
        af.v8[1] = *(const v8bf*)(aRow + kk + koff + 16);
        bf.v8[0] = *(const v8bf*)(wRow + kk + koff);
        bf.v8[1] = *(const v8bf*)(wRow + kk + koff + 16);
        acc = __builtin_amdgcn_wmma_f32_16x16x32_bf16(
            false, af.v16, false, bf.v16, (short)0, acc, false, false);
    }
    int m0 = mt * 16 + (lane >> 4) * 8;
    int n  = nt * 16 + (lane & 15);
    float bv = bias[n];
#pragma unroll
    for (int r = 0; r < 8; ++r)
        outB[(size_t)(m0 + r) * NOUTP + n] = acc[r] + bv;
}

// ---------------------------------------------------------------- LSTM pointwise
__global__ void k_lstm_cell(const float* __restrict__ gates, float* __restrict__ c,
                            __bf16* __restrict__ A) {
    int idx = blockIdx.x * blockDim.x + threadIdx.x;
    if (idx >= BATCH * HID) return;
    int b = idx >> 10, j = idx & (HID - 1);
    const float* g = gates + (size_t)b * G4H;
    float gi = g[j], gf = g[HID + j], gg = g[2 * HID + j], go = g[3 * HID + j];
    float cn = sigm(gf) * c[idx] + sigm(gi) * tanhf(gg);
    float hn = sigm(go) * tanhf(cn);
    c[idx] = cn;
    A[(size_t)b * KCAT + ZDIM + j] = f2bf(hn);   // h (bf16) for next GEMMs
}

// ---------------------------------------------------------------- stack + decoder
// One block per batch column. Replicates reference exactly, including the
// s.reshape(B, MAXT) flat reinterpretation in the pop (reads sIn[b*129 + j]).
__global__ __launch_bounds__(128) void k_stack(const float* __restrict__ outB,
                                               float* __restrict__ V,
                                               const float* __restrict__ sIn,
                                               float* __restrict__ sOut,
                                               const float* __restrict__ x,
                                               float* __restrict__ bdone,
                                               float* __restrict__ yOut,
                                               __bf16* __restrict__ A, int t) {
    int b = blockIdx.x, tid = threadIdx.x;
    __shared__ float scol[MAXT], suf[MAXT], snew[MAXT], coeff[MAXT];
    __shared__ float red[128];
    __shared__ int   redi[128];
    __shared__ float sc[3];

    for (int j = tid; j < MAXT; j += 128) scol[j] = sIn[(size_t)j * BATCH + b];
    if (tid < DS) V[((size_t)t * BATCH + b) * DS + tid] = outB[(size_t)b * NOUTP + DIN + tid];
    if (tid == 0) {
        sc[0] = sigm(outB[(size_t)b * NOUTP + DIN + DS]);       // dt
        sc[1] = sigm(outB[(size_t)b * NOUTP + DIN + DS + 1]);   // ut
    }
    __syncthreads();
    if (tid == 0) {                         // suffix sums of old column
        float run = 0.0f;
        for (int j = MAXT - 1; j >= 0; --j) { suf[j] = run; run += scol[j]; }
        sc[2] = run;                        // total
    }
    __syncthreads();
    float dt = sc[0], ut = sc[1], totOld = sc[2];

    for (int j = tid; j < MAXT; j += 128) { // pop (scrambled flat read) + push
        float sv;
        if (t > 0) {
            float p = (j < t) ? suf[j] : totOld;
            float pop = ut - p; pop = pop > 0.0f ? pop : 0.0f;
            float src = sIn[(size_t)b * MAXT + j];   // flat reshape view
            sv = src - pop; sv = sv > 0.0f ? sv : 0.0f;
        } else sv = scol[j];
        if (j == t) sv = dt;
        snew[j] = sv;
    }
    __syncthreads();
    if (tid == 0) {                         // read coefficients
        float tot = 0.0f;
        for (int j = 0; j < MAXT; ++j) tot += snew[j];
        float run = 0.0f;
        for (int j = MAXT - 1; j >= 0; --j) {
            float p = (j < t) ? run : tot;
            float a = 1.0f - p; a = a > 0.0f ? a : 0.0f;
            coeff[j] = fminf(snew[j], a);
            run += snew[j];
        }
    }
    __syncthreads();
    for (int j = tid; j < MAXT; j += 128) sOut[(size_t)j * BATCH + b] = snew[j];
    if (tid < DS) {                          // rt = sum_j coeff[j] * V[j,b,:]
        float acc = 0.0f;
        for (int j = 0; j < MAXT; ++j)
            acc += coeff[j] * V[((size_t)j * BATCH + b) * DS + tid];
        A[(size_t)b * KCAT + DIN + tid] = f2bf(acc);
    }
    if (t + 1 < MAXT) {                      // next-step x slot (zeros in decoder)
        float xv = (t + 1 < ENC_STEPS) ? x[((size_t)(t + 1) * BATCH + b) * DIN + tid] : 0.0f;
        A[(size_t)b * KCAT + tid] = f2bf(xv);
    }
    if (t >= ENC_STEPS) {                    // decoder: log-softmax + argmax + mask
        float v = outB[(size_t)b * NOUTP + tid];
        red[tid] = v; redi[tid] = tid;
        __syncthreads();
        for (int off = 64; off > 0; off >>= 1) {
            if (tid < off) {
                float v2 = red[tid + off]; int i2 = redi[tid + off];
                if (v2 > red[tid] || (v2 == red[tid] && i2 < redi[tid])) { red[tid] = v2; redi[tid] = i2; }
            }
            __syncthreads();
        }
        float maxv = red[0]; int amax = redi[0];
        __syncthreads();
        red[tid] = __expf(v - maxv);
        __syncthreads();
        for (int off = 64; off > 0; off >>= 1) {
            if (tid < off) red[tid] += red[tid + off];
            __syncthreads();
        }
        float logp = v - maxv - __logf(red[0]);
        float bd = bdone[b];
        float beg = (tid == 0) ? 1.0f : 0.0f;
        yOut[(((size_t)(t - ENC_STEPS)) * BATCH + b) * DIN + tid] = (1.0f - bd) * logp + bd * beg;
        __syncthreads();
        if (tid == 0) bdone[b] = fmaxf(bd, (amax == 2) ? 1.0f : 0.0f);
    }
}

// ---------------------------------------------------------------- host
extern "C" void kernel_launch(void* const* d_in, const int* in_sizes, int n_in,
                              void* d_out, int out_size, void* d_ws, size_t ws_size,
                              hipStream_t stream) {
    const float* x        = (const float*)d_in[0];
    const float* w_ih_enc = (const float*)d_in[1];
    const float* w_hh_enc = (const float*)d_in[2];
    const float* b_ih_enc = (const float*)d_in[3];
    const float* b_hh_enc = (const float*)d_in[4];
    const float* w_ih_dec = (const float*)d_in[5];
    const float* w_hh_dec = (const float*)d_in[6];
    const float* b_ih_dec = (const float*)d_in[7];
    const float* b_hh_dec = (const float*)d_in[8];
    const float* w_ro     = (const float*)d_in[9];
    const float* b_ro     = (const float*)d_in[10];

    char* p = (char*)d_ws;
    auto carve = [&](size_t bytes) -> char* {
        char* r = p; p += (bytes + 255) & ~(size_t)255; return r;
    };
    __bf16* WcatE = (__bf16*)carve((size_t)G4H * KCAT * 2);
    __bf16* WcatD = (__bf16*)carve((size_t)G4H * KCAT * 2);
    __bf16* WroP  = (__bf16*)carve((size_t)NOUTP * HID * 2);
    float*  biasE = (float*)carve(G4H * 4);
    float*  biasD = (float*)carve(G4H * 4);
    float*  broP  = (float*)carve(NOUTP * 4);
    __bf16* Abuf  = (__bf16*)carve((size_t)BATCH * KCAT * 2);
    float*  gates = (float*)carve((size_t)BATCH * G4H * 4);
    float*  cbuf  = (float*)carve((size_t)BATCH * HID * 4);
    float*  outB  = (float*)carve((size_t)BATCH * NOUTP * 4);
    float*  Vbuf  = (float*)carve((size_t)MAXT * BATCH * DS * 4);
    float*  sA    = (float*)carve((size_t)MAXT * BATCH * 4);
    float*  sB    = (float*)carve((size_t)MAXT * BATCH * 4);
    float*  bdone = (float*)carve(BATCH * 4);

    // --- per-call prep (deterministic, no caching) ---
    {
        int n = G4H * KCAT;
        k_build_wcat<<<(n + 255) / 256, 256, 0, stream>>>(WcatE, w_ih_enc, w_hh_enc);
        k_build_wcat<<<(n + 255) / 256, 256, 0, stream>>>(WcatD, w_ih_dec, w_hh_dec);
    }
    k_build_wro<<<(NOUTP * HID + 255) / 256, 256, 0, stream>>>(WroP, broP, w_ro, b_ro);
    k_build_bias<<<(G4H + 255) / 256, 256, 0, stream>>>(biasE, b_ih_enc, b_hh_enc);
    k_build_bias<<<(G4H + 255) / 256, 256, 0, stream>>>(biasD, b_ih_dec, b_hh_dec);

    { int n = MAXT * BATCH * DS; k_zero_f32<<<(n + 255) / 256, 256, 0, stream>>>(Vbuf, n); }
    { int n = MAXT * BATCH;      k_zero_f32<<<(n + 255) / 256, 256, 0, stream>>>(sA, n);
                                 k_zero_f32<<<(n + 255) / 256, 256, 0, stream>>>(sB, n); }
    { int n = BATCH * HID;       k_zero_f32<<<(n + 255) / 256, 256, 0, stream>>>(cbuf, n); }
    k_zero_f32<<<1, 256, 0, stream>>>(bdone, BATCH);
    { int n = BATCH * KCAT;      k_init_abuf<<<(n + 255) / 256, 256, 0, stream>>>(Abuf, x); }

    // --- sequential recurrence: 65 encoder + 64 decoder steps ---
    float* sIn = sA; float* sOut = sB;
    for (int st = 0; st < MAXT; ++st) {
        const __bf16* Wc = (st < ENC_STEPS) ? WcatE : WcatD;
        const float*  bc = (st < ENC_STEPS) ? biasE : biasD;
        k_gemm_gates<<<128, 256, 0, stream>>>(Abuf, Wc, bc, gates);           // 1024 waves
        k_lstm_cell<<<(BATCH * HID) / 256, 256, 0, stream>>>(gates, cbuf, Abuf);
        k_gemm_out<<<52, 128, 0, stream>>>(Abuf, WroP, broP, outB);           // 208 waves
        k_stack<<<BATCH, 128, 0, stream>>>(outB, Vbuf, sIn, sOut, x, bdone,
                                           (float*)d_out, Abuf, st);
        float* tmp = sIn; sIn = sOut; sOut = tmp;
    }
}
